// UNet_1_52123723104316
// MI455X (gfx1250) — matrix operand order repaired and checked by reading, not compiled, not run on hardware
//
#include <hip/hip_runtime.h>

// ---------------------------------------------------------------------------
// Types for CDNA5 WMMA (wave32).
// ---------------------------------------------------------------------------
typedef __attribute__((ext_vector_type(16))) __bf16 v16bf;
typedef __attribute__((ext_vector_type(8)))  float  v8f;

static constexpr int BN = 2;          // batch
static constexpr float BN_EPS = 1e-5f;

// float -> bf16 round-to-nearest-even (scalar fallback)
__device__ __forceinline__ unsigned short f2bf(float f) {
    unsigned int u = __float_as_uint(f);
    unsigned int r = u + 0x7fffu + ((u >> 16) & 1u);
    return (unsigned short)(r >> 16);
}

// pack two floats into one u32 of bf16 pair; prefer HW packed convert if declared
#if __has_builtin(__builtin_amdgcn_cvt_pk_bf16_f32)
typedef __attribute__((ext_vector_type(2))) __bf16 v2bf;
__device__ __forceinline__ unsigned int pk2(float lo, float hi) {
    v2bf r = __builtin_amdgcn_cvt_pk_bf16_f32(lo, hi);
    return __builtin_bit_cast(unsigned int, r);
}
#else
__device__ __forceinline__ unsigned int pk2(float lo, float hi) {
    return (unsigned int)f2bf(lo) | ((unsigned int)f2bf(hi) << 16);
}
#endif

// ---------------------------------------------------------------------------
// Pack fp32 OIHW weights into the EXACT CDNA5 B-fragment order:
//   u32[((ctile*nkt + ktile)*32 + lane)*8 + v]
// bf16 pair for cout = ctile*16 + (lane&15), k = ktile*32 + 2v + (v>=4?8:0) + (lane>>4)*8 (+0/1),
// tap-major K: k -> (tap = k>>cshift, cin = k & (Cin-1)); zero beyond K.
// ---------------------------------------------------------------------------
__global__ void pack_w_kernel(const float* __restrict__ w, unsigned int* __restrict__ out,
                              int Cin, int K, int cshift, int nkt, long total) {
    long i = (long)blockIdx.x * blockDim.x + threadIdx.x;
    if (i >= total) return;
    int v    = (int)(i & 7);
    int lane = (int)((i >> 3) & 31);
    long t   = i >> 8;                  // tile index = ctile*nkt + ktile
    int kt   = (int)(t % nkt);
    int ct   = (int)(t / nkt);
    int half = lane >> 4, n = lane & 15;
    int cout = ct * 16 + n;
    unsigned int pk = 0;
#pragma unroll
    for (int tt = 0; tt < 2; ++tt) {
        int k = kt * 32 + 2 * v + ((v >= 4) ? 8 : 0) + half * 8 + tt;
        float val = 0.f;
        if (k < K) {
            int tap = k >> cshift;
            int cin = k & (Cin - 1);
            val = w[((long)cout * Cin + cin) * 9 + tap];
        }
        pk |= (unsigned int)f2bf(val) << (16 * tt);
    }
    out[i] = pk;
}

// ---------------------------------------------------------------------------
// Implicit-GEMM 3x3 SAME conv, NHWC, via v_wmma_f32_16x16x32_bf16 (wave32).
//   A = activations (M = 16 pixels), B = pre-swizzled weights (N = 16 couts).
// Per K-step: issue 4x b128 activation loads + 2*MT b128 weight loads into
// DISTINCT fragment registers (one load clause), overlap the bf16 pack VALU
// with load latency, then fire MT WMMAs back-to-back.
// VEC=false (Cin==1, first layer only): single padded K-step, per-element decode.
// ---------------------------------------------------------------------------
template <int MT, bool VEC>
__global__ void __launch_bounds__(256)
conv3x3_wmma_kernel(const float* __restrict__ in,            // NHWC [B,H,W,Cin]
                    const unsigned int* __restrict__ wsw,    // swizzled bf16 weights
                    const float* __restrict__ bias, float* __restrict__ out, // NHWC
                    int Cin, int Cout, int H, int W, int Kpad, int cshift, int wshift) {
    const int HW   = H * W;
    const int npix = HW >> 4;
    const int nct  = Cout / (16 * MT);
    const int nkt  = Kpad >> 5;
    const int tilesPerBatch = npix * nct;
    const int totalTiles = BN * tilesPerBatch;

    int wave = blockIdx.x * (blockDim.x >> 5) + (threadIdx.x >> 5);
    if (wave >= totalTiles) return;            // wave-uniform: EXEC all-1s for WMMA

    const int lane = threadIdx.x & 31;
    const int half = lane >> 4;
    const int l16  = lane & 15;

    int b  = wave / tilesPerBatch;
    int r  = wave - b * tilesPerBatch;
    int pt = r / nct;                          // pixel tile (16 pixels)
    int ct = r - pt * nct;                     // super cout tile (MT x 16)

    const int pix = pt * 16 + l16;             // A row M = l16 (both halves same pixel)
    const int py  = pix >> wshift;
    const int pxx = pix & (W - 1);

    const float* inb = in + (size_t)b * HW * Cin;

    const unsigned int* wp[MT];
#pragma unroll
    for (int m = 0; m < MT; ++m)
        wp[m] = wsw + (((size_t)(ct * MT + m) * nkt) * 32 + lane) * 8;

    float bv[MT];
#pragma unroll
    for (int m = 0; m < MT; ++m) bv[m] = bias[(ct * MT + m) * 16 + l16];

    v8f zero = {};
    v8f acc[MT];
#pragma unroll
    for (int m = 0; m < MT; ++m) acc[m] = zero;

    union BwU { unsigned int u[8]; v16bf v; };

    if (VEC) {
        // ---- tap-outer: address/clamp/bounds hoisted out of the channel loop
        for (int tap = 0; tap < 9; ++tap) {
            int dy = (tap >= 6) ? 2 : ((tap >= 3) ? 1 : 0);
            int dx = tap - 3 * dy;
            int yy = py + dy - 1, xx = pxx + dx - 1;
            int ok = (yy >= 0) & (yy < H) & (xx >= 0) & (xx < W);
            yy = yy < 0 ? 0 : (yy >= H ? H - 1 : yy);   // clamp: load always legal
            xx = xx < 0 ? 0 : (xx >= W ? W - 1 : xx);
            unsigned int zm = ok ? 0xffffffffu : 0u;    // boundary zero, branchless
            const float* p = inb + ((size_t)(yy * W + xx)) * Cin + half * 8;
            for (int cb = 0; cb < Cin; cb += 32) {
                // -- issue all loads first (activations + MT weight fragments)
                float4 f0 = *(const float4*)(p + cb);        // cin run0: half*8 + 0..7
                float4 f1 = *(const float4*)(p + cb + 4);
                float4 f2 = *(const float4*)(p + cb + 16);   // cin run1: half*8 + 16..23
                float4 f3 = *(const float4*)(p + cb + 20);
                BwU Bw[MT];
#pragma unroll
                for (int m = 0; m < MT; ++m) {
                    *(uint4*)&Bw[m].u[0] = *(const uint4*)(wp[m]);
                    *(uint4*)&Bw[m].u[4] = *(const uint4*)(wp[m] + 4);
                    wp[m] += 256;                            // next K tile
                }
                // -- pack VALU overlaps weight-load latency
                union { unsigned int u[8]; v16bf v; } Aa;
                Aa.u[0] = pk2(f0.x, f0.y) & zm;  Aa.u[1] = pk2(f0.z, f0.w) & zm;
                Aa.u[2] = pk2(f1.x, f1.y) & zm;  Aa.u[3] = pk2(f1.z, f1.w) & zm;
                Aa.u[4] = pk2(f2.x, f2.y) & zm;  Aa.u[5] = pk2(f2.z, f2.w) & zm;
                Aa.u[6] = pk2(f3.x, f3.y) & zm;  Aa.u[7] = pk2(f3.z, f3.w) & zm;
                // -- WMMA burst
#pragma unroll
                for (int m = 0; m < MT; ++m)
                    acc[m] = __builtin_amdgcn_wmma_f32_16x16x32_bf16(
                        false, Aa.v, false, Bw[m].v, (short)0, acc[m], false, false);
            }
        }
    } else {
        // ---- Cin==1 (first layer): K=9 padded to one 32-step, per-element decode
        const int Kreal = Cin * 9;
        BwU Bw[MT];
#pragma unroll
        for (int m = 0; m < MT; ++m) {
            *(uint4*)&Bw[m].u[0] = *(const uint4*)(wp[m]);
            *(uint4*)&Bw[m].u[4] = *(const uint4*)(wp[m] + 4);
        }
        union { unsigned int u[8]; v16bf v; } Aa;
#pragma unroll
        for (int v8i = 0; v8i < 8; ++v8i) {
            unsigned int pk = 0;
#pragma unroll
            for (int t = 0; t < 2; ++t) {
                int k = 2 * v8i + ((v8i >= 4) ? 8 : 0) + half * 8 + t;
                int ok = (k < Kreal) ? 1 : 0;
                int tap = ok ? k : 0;
                int dy = (tap >= 6) ? 2 : ((tap >= 3) ? 1 : 0);
                int dx = tap - 3 * dy;
                int yy = py + dy - 1, xx = pxx + dx - 1;
                ok &= (yy >= 0) & (yy < H) & (xx >= 0) & (xx < W);
                yy = yy < 0 ? 0 : (yy >= H ? H - 1 : yy);
                xx = xx < 0 ? 0 : (xx >= W ? W - 1 : xx);
                float f = inb[(size_t)(yy * W + xx)];
                f = ok ? f : 0.f;
                pk |= (unsigned int)f2bf(f) << (16 * t);
            }
            Aa.u[v8i] = pk;
        }
#pragma unroll
        for (int m = 0; m < MT; ++m)
            acc[m] = __builtin_amdgcn_wmma_f32_16x16x32_bf16(
                false, Aa.v, false, Bw[m].v, (short)0, acc[m], false, false);
    }

    // D: VGPR ri -> pixel = pt*16 + ri + 8*half, cout = tile*16 + (lane&15): coalesced
    float* ob = out + (size_t)b * HW * Cout;
#pragma unroll
    for (int m = 0; m < MT; ++m) {
#pragma unroll
        for (int ri = 0; ri < 8; ++ri) {
            int pixel = pt * 16 + ri + half * 8;
            ob[(size_t)pixel * Cout + (ct * MT + m) * 16 + l16] = acc[m][ri] + bv[m];
        }
    }
}

// ---------------------------------------------------------------------------
// BN (training mode) on NHWC: zero partials -> one coalesced pass with LDS
// atomic accumulators -> fused finalize+scale/shift+ReLU.
// ---------------------------------------------------------------------------
__global__ void zero_kernel(float* p, int n) {
    int i = blockIdx.x * blockDim.x + threadIdx.x;
    if (i < n) p[i] = 0.f;
}

__global__ void bn_stats_kernel(const float* __restrict__ x, float* __restrict__ ps,
                                int C, long total) {
    __shared__ float sb[1024];                 // [0..C): sum, [C..2C): sumsq (C<=512)
    for (int j = threadIdx.x; j < 2 * C; j += blockDim.x) sb[j] = 0.f;
    __syncthreads();
    for (long i = (long)blockIdx.x * blockDim.x + threadIdx.x; i < total;
         i += (long)gridDim.x * blockDim.x) {
        float v = x[i];
        int c = (int)(i & (C - 1));            // NHWC: channel = low bits
        atomicAdd(&sb[c], v);
        atomicAdd(&sb[C + c], v * v);
    }
    __syncthreads();
    for (int j = threadIdx.x; j < 2 * C; j += blockDim.x) atomicAdd(&ps[j], sb[j]);
}

__global__ void bn_apply_kernel(const float* __restrict__ x, float* __restrict__ y,
                                const float* __restrict__ ps,
                                const float* __restrict__ g, const float* __restrict__ be,
                                int C, float invN, long total) {
    long i = (long)blockIdx.x * blockDim.x + threadIdx.x;
    if (i >= total) return;
    int c = (int)(i & (C - 1));
    float m  = ps[c] * invN;
    float vr = ps[C + c] * invN - m * m;
    float v = (x[i] - m) * rsqrtf(vr + BN_EPS) * g[c] + be[c];
    y[i] = v > 0.f ? v : 0.f;
}

// ---------------------------------------------------------------------------
// 2x2 max pool, stride 2, NHWC.
// ---------------------------------------------------------------------------
__global__ void maxpool_kernel(const float* __restrict__ x, float* __restrict__ y,
                               int C, int H, int W, long total) {
    long i = (long)blockIdx.x * blockDim.x + threadIdx.x;
    if (i >= total) return;
    int c = (int)(i & (C - 1));
    long p = i / C;
    int OW = W >> 1, OH = H >> 1;
    int ox = (int)(p % OW); p /= OW;
    int oy = (int)(p % OH);
    int b  = (int)(p / OH);
    const float* s = x + (((size_t)(b * H + oy * 2) * W) + ox * 2) * C + c;
    size_t rs = (size_t)W * C;
    float m0 = fmaxf(s[0], s[C]);
    float m1 = fmaxf(s[rs], s[rs + C]);
    y[i] = fmaxf(m0, m1);
}

// ---------------------------------------------------------------------------
// Bilinear resize (align_corners=True), NHWC.
// ---------------------------------------------------------------------------
__global__ void upsample_kernel(const float* __restrict__ x, float* __restrict__ y,
                                int C, int H, int W, int OH, int OW, long total) {
    long i = (long)blockIdx.x * blockDim.x + threadIdx.x;
    if (i >= total) return;
    int c = (int)(i & (C - 1));
    long p = i / C;
    int ox = (int)(p % OW); p /= OW;
    int oy = (int)(p % OH);
    int b  = (int)(p / OH);
    float sy = (OH > 1) ? (float)(H - 1) / (float)(OH - 1) : 0.f;
    float sx = (OW > 1) ? (float)(W - 1) / (float)(OW - 1) : 0.f;
    float fy = oy * sy, fx = ox * sx;
    int y0 = (int)fy, x0 = (int)fx;
    int y1 = (y0 + 1 < H) ? y0 + 1 : H - 1;
    int x1 = (x0 + 1 < W) ? x0 + 1 : W - 1;
    float wy = fy - y0, wx = fx - x0;
    const float* base = x + (size_t)b * H * W * C + c;
    float v00 = base[((size_t)y0 * W + x0) * C], v01 = base[((size_t)y0 * W + x1) * C];
    float v10 = base[((size_t)y1 * W + x0) * C], v11 = base[((size_t)y1 * W + x1) * C];
    y[i] = (v00 * (1.f - wx) + v01 * wx) * (1.f - wy) + (v10 * (1.f - wx) + v11 * wx) * wy;
}

// ---------------------------------------------------------------------------
// Channel-concat, NHWC: src [*,Cs] -> dst [*,Ct] at channel offset co.
// ---------------------------------------------------------------------------
__global__ void concat_kernel(const float* __restrict__ src, float* __restrict__ dst,
                              int Cs, int Ct, int co, long total) {
    long i = (long)blockIdx.x * blockDim.x + threadIdx.x;
    if (i >= total) return;
    int c = (int)(i & (Cs - 1));
    long p = i / Cs;
    dst[(size_t)p * Ct + co + c] = src[i];
}

// ---------------------------------------------------------------------------
// Spatially-varying 5x5 conv, NHWC; kern [B,H,W,5,5] shared over channels.
// ---------------------------------------------------------------------------
__global__ void smooth_kernel(const float* __restrict__ x, const float* __restrict__ kern,
                              float* __restrict__ y, int C, int H, int W, long total) {
    long i = (long)blockIdx.x * blockDim.x + threadIdx.x;
    if (i >= total) return;
    int c = (int)(i & (C - 1));
    long p = i / C;
    int xx = (int)(p % W); p /= W;
    int yy = (int)(p % H);
    int b  = (int)(p / H);
    const float* base = x + (size_t)b * H * W * C + c;
    const float* kp = kern + (((size_t)b * H + yy) * W + xx) * 25;
    float s = 0.f;
#pragma unroll
    for (int a = 0; a < 5; ++a) {
        int iy = yy + a - 2;
#pragma unroll
        for (int d = 0; d < 5; ++d) {
            int ix = xx + d - 2;
            float v = (iy >= 0 && iy < H && ix >= 0 && ix < W)
                          ? base[((size_t)iy * W + ix) * C] : 0.f;
            s += v * kp[a * 5 + d];
        }
    }
    y[i] = s;
}

// ---------------------------------------------------------------------------
// Final 1x1 conv, NHWC: 32 -> 1 channel (vectorized channel dot).
// ---------------------------------------------------------------------------
__global__ void conv1x1_kernel(const float* __restrict__ x, const float* __restrict__ w,
                               const float* __restrict__ bsc, float* __restrict__ y,
                               long total) {
    long i = (long)blockIdx.x * blockDim.x + threadIdx.x;
    if (i >= total) return;
    const float* p = x + i * 32;
    float s = bsc[0];
#pragma unroll
    for (int q = 0; q < 8; ++q) {
        float4 f = *(const float4*)(p + q * 4);
        s += w[q * 4] * f.x + w[q * 4 + 1] * f.y + w[q * 4 + 2] * f.z + w[q * 4 + 3] * f.w;
    }
    y[i] = s;
}

// ---------------------------------------------------------------------------
// Host orchestration. All internal activations NHWC (I/O tensors have C=1).
// ---------------------------------------------------------------------------
struct CBp { const float *w1, *b1, *g1, *be1, *w2, *b2, *g2, *be2; };
struct UPp { const float *w1, *b1, *g1, *be1; };
struct PK  { unsigned int* p; int Kpad; int cshift; };

static inline unsigned gdim(long total) { return (unsigned)((total + 255) / 256); }

extern "C" void kernel_launch(void* const* d_in, const int* in_sizes, int n_in,
                              void* d_out, int out_size, void* d_ws, size_t ws_size,
                              hipStream_t stream) {
    (void)in_sizes; (void)n_in; (void)out_size; (void)ws_size;

    const float* X  = (const float*)d_in[0];     // [2,1,256,256] (C=1 -> NHWC==NCHW)
    const float* F4 = (const float*)d_in[1];     // [2,64,64,5,5]
    const float* F8 = (const float*)d_in[2];     // [2,32,32,5,5]
    int idx = 3;
    auto nextCB = [&]() { CBp p;
        p.w1 = (const float*)d_in[idx++]; p.b1 = (const float*)d_in[idx++];
        p.g1 = (const float*)d_in[idx++]; p.be1 = (const float*)d_in[idx++];
        p.w2 = (const float*)d_in[idx++]; p.b2 = (const float*)d_in[idx++];
        p.g2 = (const float*)d_in[idx++]; p.be2 = (const float*)d_in[idx++];
        return p; };
    auto nextUP = [&]() { UPp p;
        p.w1 = (const float*)d_in[idx++]; p.b1 = (const float*)d_in[idx++];
        p.g1 = (const float*)d_in[idx++]; p.be1 = (const float*)d_in[idx++];
        return p; };
    CBp Conv1 = nextCB(), Conv2 = nextCB(), Conv3 = nextCB(), Conv4 = nextCB(), Conv5 = nextCB();
    UPp Up5 = nextUP(); CBp Upc5 = nextCB();
    UPp Up4 = nextUP(); CBp Upc4 = nextCB();
    UPp Up3 = nextUP(); CBp Upc3 = nextCB();
    UPp Up2 = nextUP(); CBp Upc2 = nextCB();
    const float* W1x1 = (const float*)d_in[idx++];
    const float* B1x1 = (const float*)d_in[idx++];

    size_t cur = 0;
    auto allocF = [&](size_t elems) {
        float* p = (float*)((char*)d_ws + cur);
        cur = (cur + elems * sizeof(float) + 255) & ~(size_t)255;
        return p; };
    auto allocU = [&](size_t elems) {
        unsigned int* p = (unsigned int*)((char*)d_ws + cur);
        cur = (cur + elems * sizeof(unsigned int) + 255) & ~(size_t)255;
        return p; };

    float* x1    = allocF((size_t)BN * 32 * 256 * 256);
    float* x2    = allocF((size_t)BN * 64 * 128 * 128);
    float* x3    = allocF((size_t)BN * 128 * 64 * 64);
    float* x4    = allocF((size_t)BN * 256 * 32 * 32);
    float* x5    = allocF((size_t)BN * 512 * 16 * 16);
    float* p1    = allocF((size_t)BN * 32 * 128 * 128);
    float* p2    = allocF((size_t)BN * 64 * 64 * 64);
    float* p3    = allocF((size_t)BN * 128 * 32 * 32);
    float* p4    = allocF((size_t)BN * 256 * 16 * 16);
    float* d5up  = allocF((size_t)BN * 256 * 32 * 32);
    float* d5cat = allocF((size_t)BN * 512 * 32 * 32);
    float* smA5  = allocF((size_t)BN * 512 * 32 * 32);
    float* smB5  = allocF((size_t)BN * 512 * 32 * 32);
    float* d5    = allocF((size_t)BN * 256 * 32 * 32);
    float* d4up  = allocF((size_t)BN * 128 * 64 * 64);
    float* d4cat = allocF((size_t)BN * 256 * 64 * 64);
    float* smA4  = allocF((size_t)BN * 256 * 64 * 64);
    float* smB4  = allocF((size_t)BN * 256 * 64 * 64);
    float* d4    = allocF((size_t)BN * 128 * 64 * 64);
    float* d3up  = allocF((size_t)BN * 64 * 128 * 128);
    float* d3cat = allocF((size_t)BN * 128 * 128 * 128);
    float* d3    = allocF((size_t)BN * 64 * 128 * 128);
    float* d2up  = allocF((size_t)BN * 32 * 256 * 256);
    float* d2    = allocF((size_t)BN * 32 * 256 * 256);
    float* tA    = allocF((size_t)BN * 32 * 256 * 256);
    float* tB    = allocF((size_t)BN * 32 * 256 * 256);
    float* tU    = allocF((size_t)BN * 64 * 256 * 256);
    float* psb   = allocF(1024);

    auto pack = [&](const float* w, int Cout, int Cin) {
        int K = Cin * 9, Kpad = ((K + 31) / 32) * 32;
        PK pk; pk.Kpad = Kpad;
        pk.cshift = __builtin_ctz((unsigned)Cin);
        int nkt = Kpad / 32;
        long tot = (long)(Cout / 16) * nkt * 32 * 8;
        pk.p = allocU((size_t)tot);
        pack_w_kernel<<<gdim(tot), 256, 0, stream>>>(w, pk.p, Cin, K, pk.cshift, nkt, tot);
        return pk; };

    PK k1a = pack(Conv1.w1, 32, 1),    k1b = pack(Conv1.w2, 32, 32);
    PK k2a = pack(Conv2.w1, 64, 32),   k2b = pack(Conv2.w2, 64, 64);
    PK k3a = pack(Conv3.w1, 128, 64),  k3b = pack(Conv3.w2, 128, 128);
    PK k4a = pack(Conv4.w1, 256, 128), k4b = pack(Conv4.w2, 256, 256);
    PK k5a = pack(Conv5.w1, 512, 256), k5b = pack(Conv5.w2, 512, 512);
    PK ku5 = pack(Up5.w1, 256, 512);
    PK kc5a = pack(Upc5.w1, 256, 512), kc5b = pack(Upc5.w2, 256, 256);
    PK ku4 = pack(Up4.w1, 128, 256);
    PK kc4a = pack(Upc4.w1, 128, 256), kc4b = pack(Upc4.w2, 128, 128);
    PK ku3 = pack(Up3.w1, 64, 128);
    PK kc3a = pack(Upc3.w1, 64, 128),  kc3b = pack(Upc3.w2, 64, 64);
    PK ku2 = pack(Up2.w1, 32, 64);
    PK kc2a = pack(Upc2.w1, 32, 64),   kc2b = pack(Upc2.w2, 32, 32);

    auto conv = [&](const float* in, PK pk, const float* bias, float* out,
                    int Cin, int Cout, int H, int W) {
        int wshift = __builtin_ctz((unsigned)W);
        if (Cout >= 64) {            // all Cout>=64 layers have Cin>=32
            long waves = (long)BN * ((H * W) / 16) * (Cout / 64);
            conv3x3_wmma_kernel<4, true><<<(unsigned)((waves + 7) / 8), 256, 0, stream>>>(
                in, pk.p, bias, out, Cin, Cout, H, W, pk.Kpad, pk.cshift, wshift);
        } else if (Cin >= 32) {
            long waves = (long)BN * ((H * W) / 16) * (Cout / 32);
            conv3x3_wmma_kernel<2, true><<<(unsigned)((waves + 7) / 8), 256, 0, stream>>>(
                in, pk.p, bias, out, Cin, Cout, H, W, pk.Kpad, pk.cshift, wshift);
        } else {                     // Conv1 first layer: Cin==1
            long waves = (long)BN * ((H * W) / 16) * (Cout / 32);
            conv3x3_wmma_kernel<2, false><<<(unsigned)((waves + 7) / 8), 256, 0, stream>>>(
                in, pk.p, bias, out, Cin, Cout, H, W, pk.Kpad, pk.cshift, wshift);
        }
    };
    auto bnrelu = [&](const float* x, float* y, const float* g, const float* be,
                      int C, int H, int W) {
        long total = (long)BN * C * H * W;
        zero_kernel<<<4, 256, 0, stream>>>(psb, 2 * C);
        long sb = (total + 255) / 256; if (sb > 512) sb = 512;
        bn_stats_kernel<<<(unsigned)sb, 256, 0, stream>>>(x, psb, C, total);
        float invN = 1.f / (float)(BN * H * W);
        bn_apply_kernel<<<gdim(total), 256, 0, stream>>>(x, y, psb, g, be, C, invN, total);
    };
    auto cblock = [&](const float* in, const CBp& p, PK pk1, PK pk2, float* out,
                      int Cin, int Cout, int H, int W) {
        conv(in, pk1, p.b1, tA, Cin, Cout, H, W);
        bnrelu(tA, tB, p.g1, p.be1, Cout, H, W);
        conv(tB, pk2, p.b2, tA, Cout, Cout, H, W);
        bnrelu(tA, out, p.g2, p.be2, Cout, H, W);
    };
    auto pool = [&](const float* in, float* out, int C, int H, int W) {
        long tot = (long)BN * C * (H / 2) * (W / 2);
        maxpool_kernel<<<gdim(tot), 256, 0, stream>>>(in, out, C, H, W, tot);
    };
    auto upc = [&](const float* in, const UPp& p, PK pk, float* out,
                   int Cin, int Cout, int H, int W) {
        int OH = 2 * H, OW = 2 * W;
        long tot = (long)BN * Cin * OH * OW;
        upsample_kernel<<<gdim(tot), 256, 0, stream>>>(in, tU, Cin, H, W, OH, OW, tot);
        conv(tU, pk, p.b1, tA, Cin, Cout, OH, OW);
        bnrelu(tA, out, p.g1, p.be1, Cout, OH, OW);
    };
    auto cat = [&](const float* src, float* dst, int Cs, int Ct, int co, int HW) {
        long tot = (long)BN * HW * Cs;
        concat_kernel<<<gdim(tot), 256, 0, stream>>>(src, dst, Cs, Ct, co, tot);
    };
    auto smooth = [&](const float* x, const float* kern, float* y, int C, int H, int W) {
        long tot = (long)BN * C * H * W;
        smooth_kernel<<<gdim(tot), 256, 0, stream>>>(x, kern, y, C, H, W, tot);
    };

    // ---- encoder ----
    cblock(X, Conv1, k1a, k1b, x1, 1, 32, 256, 256);
    pool(x1, p1, 32, 256, 256);  cblock(p1, Conv2, k2a, k2b, x2, 32, 64, 128, 128);
    pool(x2, p2, 64, 128, 128);  cblock(p2, Conv3, k3a, k3b, x3, 64, 128, 64, 64);
    pool(x3, p3, 128, 64, 64);   cblock(p3, Conv4, k4a, k4b, x4, 128, 256, 32, 32);
    pool(x4, p4, 256, 32, 32);   cblock(p4, Conv5, k5a, k5b, x5, 256, 512, 16, 16);

    // ---- decoder level 5 (32x32) ----
    upc(x5, Up5, ku5, d5up, 512, 256, 16, 16);
    cat(x4, d5cat, 256, 512, 0, 32 * 32);
    cat(d5up, d5cat, 256, 512, 256, 32 * 32);
    smooth(d5cat, F8, smA5, 512, 32, 32);
    smooth(smA5, F8, smB5, 512, 32, 32);
    cblock(smB5, Upc5, kc5a, kc5b, d5, 512, 256, 32, 32);

    // ---- decoder level 4 (64x64) ----
    upc(d5, Up4, ku4, d4up, 256, 128, 32, 32);
    cat(x3, d4cat, 128, 256, 0, 64 * 64);
    cat(d4up, d4cat, 128, 256, 128, 64 * 64);
    smooth(d4cat, F4, smA4, 256, 64, 64);
    smooth(smA4, F4, smB4, 256, 64, 64);
    cblock(smB4, Upc4, kc4a, kc4b, d4, 256, 128, 64, 64);

    // ---- decoder level 3 (128x128) ----
    upc(d4, Up3, ku3, d3up, 128, 64, 64, 64);
    cat(x2, d3cat, 64, 128, 0, 128 * 128);
    cat(d3up, d3cat, 64, 128, 64, 128 * 128);
    cblock(d3cat, Upc3, kc3a, kc3b, d3, 128, 64, 128, 128);

    // ---- decoder level 2 (256x256, no skip) ----
    upc(d3, Up2, ku2, d2up, 64, 32, 128, 128);
    cblock(d2up, Upc2, kc2a, kc2b, d2, 32, 32, 256, 256);

    // ---- final 1x1 conv ----
    long tot = (long)BN * 256 * 256;
    conv1x1_kernel<<<gdim(tot), 256, 0, stream>>>(d2, W1x1, B1x1, (float*)d_out, tot);
}
